// FastAttention_14130442404167
// MI455X (gfx1250) — matrix operand order repaired and checked by reading, not compiled, not run on hardware
//
#include <hip/hip_runtime.h>
#include <hip/hip_bf16.h>

#define BATCH 4
#define SEQ   4096
#define HEADS 16
#define DHEAD 64
#define MFEAT 256
#define NHEAD (BATCH*HEADS)
#define CPAD  80           // 64 V cols + 1 ones col, padded to 5x16
#define QCH   8            // phase-2 L chunks per head
#define ROWS2 (SEQ/QCH)    // 512
#define SUB   128          // rows per LDS sub-tile
#define NORMC 0.35355339059327373f   // 64^-0.25
#define INVSQM 0.0625f               // 1/sqrt(256)
#define EPSV  1e-6f

typedef __attribute__((ext_vector_type(16))) __bf16 v16bf;
typedef __attribute__((ext_vector_type(8)))  __bf16 v8bf;
typedef __attribute__((ext_vector_type(8)))  float  v8f;

__device__ __forceinline__ v8f zero8() {
  v8f z;
#pragma unroll
  for (int i = 0; i < 8; ++i) z[i] = 0.0f;
  return z;
}

__device__ __forceinline__ v8f wmma_bf16(v16bf a, v16bf b, v8f c) {
  // D = A(16x32 bf16) x B(32x16 bf16) + C(16x16 f32)
  return __builtin_amdgcn_wmma_f32_16x16x32_bf16(false, a, false, b, (short)0, c,
                                                 false, false);
}

// A fragment (16x32 bf16) from row-major LDS [row][K], stride in elements.
// Layout: M = lane&15; K = kb + 16*hi + j (contiguous, matches load_b_frag)
__device__ __forceinline__ v16bf load_a_frag(const __bf16* base, int row, int stride,
                                             int kb, int lane) {
  const __bf16* p = base + (size_t)row * stride + kb + ((lane & 16) ? 16 : 0);
  union { v16bf v; v8bf h[2]; } u;
  u.h[0] = *(const v8bf*)(p);       // K = kb+16*hi .. +7
  u.h[1] = *(const v8bf*)(p + 8);   // K = kb+16*hi+8 .. +7
  return u.v;
}

// B fragment (32x16 bf16) from LDS stored as [N][K] (K contiguous), stride in elements.
// Layout: N = ncol + (lane&15); K = kb + 16*(lane>=16) + j  -> one contiguous 32B read.
__device__ __forceinline__ v16bf load_b_frag(const __bf16* base, int ncol, int stride,
                                             int kb, int lane) {
  const __bf16* p = base + (size_t)(ncol + (lane & 15)) * stride + kb +
                    ((lane & 16) ? 16 : 0);
  return *(const v16bf*)(p);
}

// ---------------------------------------------------------------------------
// Phase 1: per (head, L-chunk): K' = exp(x@projT - h)/16 ; partial buf1 = K'^T @ [V|1]
// LDS: proj 32K | x 16K | kpT 64K | cT 20K | hp 1K | h 0.5K  = 136704 B
// ---------------------------------------------------------------------------
__global__ __launch_bounds__(256) void k1_kbuf(const float* __restrict__ ks,
                                               const float* __restrict__ vs,
                                               const float* __restrict__ proj,
                                               float* __restrict__ partial, int S) {
  extern __shared__ v16bf smem_v[];
  char* smem = (char*)smem_v;
  __bf16* lds_proj = (__bf16*)(smem);            // [256][64]
  __bf16* lds_x    = (__bf16*)(smem + 32768);    // [128][64]
  __bf16* lds_kp   = (__bf16*)(smem + 49152);    // [256][128]  (K'^T, K-contig in l)
  __bf16* lds_c    = (__bf16*)(smem + 114688);   // [80][128]   ([V|1]^T, K-contig in l)
  float*  lds_hp   = (float*)(smem + 135168);    // [256]
  float*  lds_h    = (float*)(smem + 136192);    // [128]

  const int tid  = threadIdx.x;
  const int lane = tid & 31;
  const int wv   = tid >> 5;
  const int lo   = lane & 15;
  const int hi16 = (lane >> 4) & 1;

  const int n  = blockIdx.x / S;
  const int s  = blockIdx.x % S;
  const int bb = n >> 4;
  const int hh = n & 15;
  const int rows_per_chunk = SEQ / S;
  const int l0 = s * rows_per_chunk;

  for (int i = tid; i < MFEAT * DHEAD; i += 256)
    lds_proj[i] = (__bf16)proj[i];
  for (int i = tid; i < 16 * SUB; i += 256) {
    int c = 64 + (i >> 7), l = i & 127;
    lds_c[c * SUB + l] = (__bf16)((c == 64) ? 1.0f : 0.0f);
  }

  v8f acc2[2][5];
#pragma unroll
  for (int a = 0; a < 2; ++a)
#pragma unroll
    for (int b = 0; b < 5; ++b) acc2[a][b] = zero8();

  __syncthreads();

  for (int sub = 0; sub < rows_per_chunk / SUB; ++sub) {
    const int lbase = l0 + sub * SUB;
    {
      const int lr = tid >> 1;
      const int d0 = (tid & 1) * 32;
      const size_t rowoff =
          (((size_t)bb * SEQ + (size_t)(lbase + lr)) * HEADS + hh) * DHEAD + d0;
      const float* kp = ks + rowoff;
      const float* vp = vs + rowoff;
      float ss = 0.f;
#pragma unroll
      for (int i = 0; i < 32; i += 4) {
        float4 f = *(const float4*)(kp + i);
        float x0 = f.x * NORMC, x1 = f.y * NORMC, x2 = f.z * NORMC, x3 = f.w * NORMC;
        ss += x0 * x0 + x1 * x1 + x2 * x2 + x3 * x3;
        lds_x[lr * 64 + d0 + i + 0] = (__bf16)x0;
        lds_x[lr * 64 + d0 + i + 1] = (__bf16)x1;
        lds_x[lr * 64 + d0 + i + 2] = (__bf16)x2;
        lds_x[lr * 64 + d0 + i + 3] = (__bf16)x3;
      }
      lds_hp[tid] = ss;
#pragma unroll
      for (int i = 0; i < 32; i += 4) {
        float4 f = *(const float4*)(vp + i);
        lds_c[(d0 + i + 0) * SUB + lr] = (__bf16)f.x;
        lds_c[(d0 + i + 1) * SUB + lr] = (__bf16)f.y;
        lds_c[(d0 + i + 2) * SUB + lr] = (__bf16)f.z;
        lds_c[(d0 + i + 3) * SUB + lr] = (__bf16)f.w;
      }
    }
    __syncthreads();
    if (tid < SUB) lds_h[tid] = 0.5f * (lds_hp[2 * tid] + lds_hp[2 * tid + 1]);
    __syncthreads();

    // GEMM1: x(128x64) @ projT(64x256), exp, store K'^T tiles
    {
      float hv[8];
#pragma unroll
      for (int r = 0; r < 8; ++r) hv[r] = lds_h[wv * 16 + hi16 * 8 + r];
      const v16bf a0 = load_a_frag(lds_x, wv * 16 + lo, 64, 0, lane);
      const v16bf a1 = load_a_frag(lds_x, wv * 16 + lo, 64, 32, lane);
#pragma unroll
      for (int mt = 0; mt < 16; ++mt) {
        v8f acc = zero8();
        v16bf b0 = load_b_frag(lds_proj, mt * 16, 64, 0, lane);
        v16bf b1 = load_b_frag(lds_proj, mt * 16, 64, 32, lane);
        acc = wmma_bf16(a0, b0, acc);
        acc = wmma_bf16(a1, b1, acc);
        v8bf pack;
#pragma unroll
        for (int r = 0; r < 8; ++r)
          pack[r] = (__bf16)(__expf(acc[r] - hv[r]) * INVSQM);
        const int m = mt * 16 + lo;
        *(v8bf*)(lds_kp + m * SUB + wv * 16 + hi16 * 8) = pack;  // 8 contiguous l
      }
    }
    __syncthreads();

    // GEMM2: buf1(256x80) += K'^T(256x128) @ C(128x80); wave -> mfTiles {2w,2w+1}
#pragma unroll
    for (int kk = 0; kk < SUB / 32; ++kk) {
      const v16bf a0 = load_a_frag(lds_kp, (2 * wv + 0) * 16 + lo, SUB, kk * 32, lane);
      const v16bf a1 = load_a_frag(lds_kp, (2 * wv + 1) * 16 + lo, SUB, kk * 32, lane);
#pragma unroll
      for (int ct = 0; ct < 5; ++ct) {
        const v16bf b = load_b_frag(lds_c, ct * 16, SUB, kk * 32, lane);
        acc2[0][ct] = wmma_bf16(a0, b, acc2[0][ct]);
        acc2[1][ct] = wmma_bf16(a1, b, acc2[1][ct]);
      }
    }
    __syncthreads();
  }

  float* outp = partial + ((size_t)n * S + s) * MFEAT * CPAD;
#pragma unroll
  for (int mt = 0; mt < 2; ++mt) {
    const int mf = (2 * wv + mt) * 16 + hi16 * 8;
#pragma unroll
    for (int ct = 0; ct < 5; ++ct) {
      const int c = ct * 16 + lo;
#pragma unroll
      for (int r = 0; r < 8; ++r)
        outp[(size_t)(mf + r) * CPAD + c] = acc2[mt][ct][r];
    }
  }
}

// ---------------------------------------------------------------------------
// Deterministic reduction of phase-1 partials
// ---------------------------------------------------------------------------
__global__ __launch_bounds__(256) void k_reduce(const float* __restrict__ partial,
                                                float* __restrict__ buf1, int S) {
  const int n = blockIdx.x;
  const float* p = partial + (size_t)n * S * MFEAT * CPAD;
  float* o = buf1 + (size_t)n * MFEAT * CPAD;
  for (int i = threadIdx.x; i < MFEAT * CPAD; i += 256) {
    float v = 0.f;
    for (int s = 0; s < S; ++s) v += p[(size_t)s * MFEAT * CPAD + i];
    o[i] = v;
  }
}

// ---------------------------------------------------------------------------
// Phase 2: Q' = exp(x@projT - h)/16 ; buf2 = Q' @ buf1 ; out = num / max(den,eps)
// LDS: proj 32K | x 16K | qp 64K | b1T 40K | hp 1K | h 0.5K | den 0.5K = 157696 B
// ---------------------------------------------------------------------------
__global__ __launch_bounds__(256) void k2_qout(const float* __restrict__ qs,
                                               const float* __restrict__ buf1,
                                               const float* __restrict__ proj,
                                               float* __restrict__ outp) {
  extern __shared__ v16bf smem_v[];
  char* smem = (char*)smem_v;
  __bf16* lds_proj = (__bf16*)(smem);            // [256][64]
  __bf16* lds_x    = (__bf16*)(smem + 32768);    // [128][64]
  __bf16* lds_qp   = (__bf16*)(smem + 49152);    // [128][256]  (Q', K-contig in m)
  __bf16* lds_b1   = (__bf16*)(smem + 114688);   // [80][256]   (buf1^T, K-contig in m)
  float*  lds_hp   = (float*)(smem + 155648);    // [256]
  float*  lds_h    = (float*)(smem + 156672);    // [128]
  float*  lds_den  = (float*)(smem + 157184);    // [128]

  const int tid  = threadIdx.x;
  const int lane = tid & 31;
  const int wv   = tid >> 5;
  const int lo   = lane & 15;
  const int hi16 = (lane >> 4) & 1;

  const int n  = blockIdx.x / QCH;
  const int qc = blockIdx.x % QCH;
  const int bb = n >> 4;
  const int hh = n & 15;
  const int l0 = qc * ROWS2;

  for (int i = tid; i < MFEAT * DHEAD; i += 256)
    lds_proj[i] = (__bf16)proj[i];
  const float* b1 = buf1 + (size_t)n * MFEAT * CPAD;
  for (int i = tid; i < MFEAT * CPAD; i += 256) {
    int m = i / CPAD, c = i % CPAD;
    lds_b1[c * MFEAT + m] = (__bf16)b1[i];
  }
  __syncthreads();

  for (int sub = 0; sub < ROWS2 / SUB; ++sub) {
    const int lbase = l0 + sub * SUB;
    {
      const int lr = tid >> 1;
      const int d0 = (tid & 1) * 32;
      const float* qp =
          qs + (((size_t)bb * SEQ + (size_t)(lbase + lr)) * HEADS + hh) * DHEAD + d0;
      float ss = 0.f;
#pragma unroll
      for (int i = 0; i < 32; i += 4) {
        float4 f = *(const float4*)(qp + i);
        float x0 = f.x * NORMC, x1 = f.y * NORMC, x2 = f.z * NORMC, x3 = f.w * NORMC;
        ss += x0 * x0 + x1 * x1 + x2 * x2 + x3 * x3;
        lds_x[lr * 64 + d0 + i + 0] = (__bf16)x0;
        lds_x[lr * 64 + d0 + i + 1] = (__bf16)x1;
        lds_x[lr * 64 + d0 + i + 2] = (__bf16)x2;
        lds_x[lr * 64 + d0 + i + 3] = (__bf16)x3;
      }
      lds_hp[tid] = ss;
    }
    __syncthreads();
    if (tid < SUB) lds_h[tid] = 0.5f * (lds_hp[2 * tid] + lds_hp[2 * tid + 1]);
    __syncthreads();

    // GEMM1 -> Q' tiles stored row-major [l][m]
    {
      float hv[8];
#pragma unroll
      for (int r = 0; r < 8; ++r) hv[r] = lds_h[wv * 16 + hi16 * 8 + r];
      const v16bf a0 = load_a_frag(lds_x, wv * 16 + lo, 64, 0, lane);
      const v16bf a1 = load_a_frag(lds_x, wv * 16 + lo, 64, 32, lane);
#pragma unroll
      for (int mt = 0; mt < 16; ++mt) {
        v8f acc = zero8();
        v16bf b0 = load_b_frag(lds_proj, mt * 16, 64, 0, lane);
        v16bf b1f = load_b_frag(lds_proj, mt * 16, 64, 32, lane);
        acc = wmma_bf16(a0, b0, acc);
        acc = wmma_bf16(a1, b1f, acc);
        const int m = mt * 16 + lo;
#pragma unroll
        for (int r = 0; r < 8; ++r)
          lds_qp[(wv * 16 + hi16 * 8 + r) * MFEAT + m] =
              (__bf16)(__expf(acc[r] - hv[r]) * INVSQM);
      }
    }
    __syncthreads();

    // GEMM3: buf2(128x80) = Q'(128x256) @ buf1(256x80); wave -> rowTile wv
    v8f acc3[5];
#pragma unroll
    for (int ct = 0; ct < 5; ++ct) acc3[ct] = zero8();
#pragma unroll
    for (int kk = 0; kk < MFEAT / 32; ++kk) {
      const v16bf a = load_a_frag(lds_qp, wv * 16 + lo, MFEAT, kk * 32, lane);
#pragma unroll
      for (int ct = 0; ct < 5; ++ct) {
        const v16bf b = load_b_frag(lds_b1, ct * 16, MFEAT, kk * 32, lane);
        acc3[ct] = wmma_bf16(a, b, acc3[ct]);
      }
    }

    // den lives in tile ct=4, N-col 0 (global col 64): lanes with lo==0
    if (lo == 0) {
#pragma unroll
      for (int r = 0; r < 8; ++r) lds_den[wv * 16 + hi16 * 8 + r] = acc3[4][r];
    }
    __syncthreads();
    float dv[8];
#pragma unroll
    for (int r = 0; r < 8; ++r) {
      float dnv = lds_den[wv * 16 + hi16 * 8 + r];
      dv[r] = 1.0f / ((dnv < EPSV) ? EPSV : dnv);
    }
    const int lrow = lbase + wv * 16 + hi16 * 8;
#pragma unroll
    for (int ct = 0; ct < 4; ++ct) {
      const int d = ct * 16 + lo;
#pragma unroll
      for (int r = 0; r < 8; ++r) {
        const int l = lrow + r;
        outp[(((size_t)bb * SEQ + l) * HEADS + hh) * DHEAD + d] = acc3[ct][r] * dv[r];
      }
    }
    __syncthreads();
  }
}

extern "C" void kernel_launch(void* const* d_in, const int* in_sizes, int n_in,
                              void* d_out, int out_size, void* d_ws, size_t ws_size,
                              hipStream_t stream) {
  (void)in_sizes; (void)n_in; (void)out_size;
  const float* qs   = (const float*)d_in[0];
  const float* ks   = (const float*)d_in[1];
  const float* vs   = (const float*)d_in[2];
  const float* proj = (const float*)d_in[3];
  float* out = (float*)d_out;

  const size_t per = (size_t)MFEAT * CPAD;  // floats per head buf1
  // Prefer 4-way L split in phase 1 (256 blocks) with deterministic reduce;
  // fall back to 1 chunk/head if workspace is small.
  int S = 4;
  if (ws_size < ((size_t)NHEAD * 4 * per + (size_t)NHEAD * per) * sizeof(float)) S = 1;

  float* partial = (float*)d_ws;
  float* buf1 = (S > 1) ? (partial + (size_t)NHEAD * S * per) : partial;

  const int SMEM1 = 136704;
  const int SMEM2 = 157696;
  (void)hipFuncSetAttribute((const void*)k1_kbuf,
                            hipFuncAttributeMaxDynamicSharedMemorySize, SMEM1);
  (void)hipFuncSetAttribute((const void*)k2_qout,
                            hipFuncAttributeMaxDynamicSharedMemorySize, SMEM2);

  k1_kbuf<<<NHEAD * S, 256, SMEM1, stream>>>(ks, vs, proj, partial, S);
  if (S > 1) k_reduce<<<NHEAD, 256, 0, stream>>>(partial, buf1, S);
  k2_qout<<<NHEAD * QCH, 256, SMEM2, stream>>>(qs, buf1, proj, out);
}
